// PosteriorEncoder_24842090840721
// MI455X (gfx1250) — matrix operand order repaired and checked by reading, not compiled, not run on hardware
//
#include <hip/hip_runtime.h>
#include <math.h>

typedef __attribute__((ext_vector_type(16))) _Float16 v16h;
typedef __attribute__((ext_vector_type(8)))  _Float16 v8h;
typedef __attribute__((ext_vector_type(8)))  float    v8f;
typedef __attribute__((ext_vector_type(4)))  unsigned int v4u;
typedef __attribute__((ext_vector_type(8)))  int v8i;
typedef __attribute__((ext_vector_type(4)))  int v4i;

#define WMMA16(A, Bf, C) __builtin_amdgcn_wmma_f32_16x16x32_f16(false, (A), false, (Bf), (short)0, (C), false, false)

namespace {
constexpr int kT    = 2048;   // time length
constexpr int kTP   = kT + 4; // f16 h buffer rows incl. +/-2 zero pad
constexpr int kH    = 192;    // hidden channels
constexpr int kTT   = 64;     // time tile per block
constexpr int kKS   = kH / 32;   // 6 K-slices of 32 for K=192
constexpr int kKSpre = 17;       // ceil(513/32) -> K padded to 544
constexpr int kTileBytes = 68 * kH * 2;           // 26112 B h tile
constexpr int kSliceB    = kTileBytes / 12;       // 2176 B per wave
constexpr int kSliceU    = kSliceB / 8;           // 272 8-byte units
}

__device__ __forceinline__ v8f vzero8() { v8f z = {}; return z; }

// one 16-byte async global->LDS copy (tracked by ASYNCcnt)
__device__ __forceinline__ void async_b128(unsigned lds_off, const void* gptr) {
  asm volatile("global_load_async_to_lds_b128 %0, %1, off"
               :: "v"(lds_off), "v"(gptr) : "memory");
}
__device__ __forceinline__ void wait_async0() {
  asm volatile("s_wait_asynccnt 0x0" ::: "memory");
}

// TDM: 1D contiguous copy of `units` 8-byte elements global->LDS (per-wave)
__device__ __forceinline__ void tdm_copy_1d(unsigned lds_addr, const void* gptr,
                                            unsigned units) {
  unsigned long long ga = (unsigned long long)(size_t)gptr;
  v4u g0 = { 1u,                                   // count=1 (valid), load
             lds_addr,                             // lds byte address
             (unsigned)ga,                         // global addr [31:0]
             ((unsigned)(ga >> 32) & 0x1FFFFFFu) | (2u << 30) }; // addr[56:32] | type=2
  v8i g1 = { (int)(3u << 16),                      // data_size = 8 bytes
             (int)((units & 0xFFFFu) << 16),       // tensor_dim0 [15:0] -> bits 63:48
             (int)((units >> 16) | (1u << 16)),    // tensor_dim0 hi | tensor_dim1=1
             (int)(units << 16),                   // tile_dim0 -> bits 127:112
             1,                                    // tile_dim1=1, tile_dim2=0
             (int)units,                           // tensor_dim0_stride lo
             0, 0 };
  v4i gz = {};
  v8i gz8 = {};
  __builtin_amdgcn_tensor_load_to_lds(g0, g1, gz, gz, gz8, 0);
}
__device__ __forceinline__ void wait_tensor0() {
  __builtin_amdgcn_s_wait_tensorcnt(0);
}

// ---------------------------------------------------------------------------
// Convert f32 weights into WMMA A-fragment-ready f16 layout.
// Fragment = 512 halfs: element e -> lane = e>>4, h = e&15.
// A layout (16x32 f16): m = mt*16 + (lane&15),
//   koff = h + (lane>=16 ? 16 : 0), k = ks*32 + koff.
// ---------------------------------------------------------------------------
__global__ void prep_frags(const float* __restrict__ src,
                           _Float16* __restrict__ dst, int mode) {
  int frag = blockIdx.x;
  int e    = threadIdx.x;          // 0..511
  int lane = e >> 4;
  int h    = e & 15;
  int koff = h + ((lane >= 16) ? 16 : 0);
  int ml   = lane & 15;
  float val = 0.f;
  if (mode == 0) {                 // in_w (L,384,192,5): frag=((l*5+d)*24+mt)*6+ks
    int ks = frag % 6; int t1 = frag / 6;
    int mt = t1 % 24;  int t2 = t1 / 24;
    int d  = t2 % 5;   int l  = t2 / 5;
    int m = mt * 16 + ml; int c = ks * 32 + koff;
    val = src[(size_t)((l * 384 + m) * 192 + c) * 5 + d];
  } else if (mode == 1) {          // rs_w (L,384,192): frag=(l*24+mt)*6+ks
    int ks = frag % 6; int t1 = frag / 6;
    int mt = t1 % 24;  int l  = t1 / 24;
    int m = mt * 16 + ml; int c = ks * 32 + koff;
    val = src[(size_t)(l * 384 + m) * 192 + c];
  } else if (mode == 2) {          // post_w (384,192): frag=mt*6+ks
    int ks = frag % 6; int mt = frag / 6;
    val = src[(size_t)(mt * 16 + ml) * 192 + ks * 32 + koff];
  } else {                         // pre_w (192,513): frag=mt*17+ks, zero pad
    int ks = frag % 17; int mt = frag / 17;
    int c = ks * 32 + koff;
    val = (c < 513) ? src[(size_t)(mt * 16 + ml) * 513 + c] : 0.f;
  }
  dst[(size_t)frag * 512 + e] = (_Float16)val;
}

// ---------------------------------------------------------------------------
// Pre conv: h0[b][t][c] = (sum_ci pre_w[c][ci]*x[b][ci][t] + pre_b[c]) * mask
// GEMM M=192 (12 m-tiles, one per wave), K=544 (17 slices), N=64 per block.
// Also: zero skip, write f16 copy of h0, zero the halo pads of both h16 bufs.
// ---------------------------------------------------------------------------
__global__ void __launch_bounds__(384)
pre_conv(const float* __restrict__ x, const int* __restrict__ xlen,
         const _Float16* __restrict__ fragPre, const float* __restrict__ pre_b,
         float* __restrict__ h0, float* __restrict__ skip,
         _Float16* __restrict__ h16a, _Float16* __restrict__ h16b) {
  __shared__ __align__(32) _Float16 xT[64 * 40]; // time-major 32-ch slice, stride 40
  int b  = blockIdx.y;
  int t0 = blockIdx.x * kTT;
  int tid  = threadIdx.x;
  int lane = tid & 31;
  int w    = tid >> 5;       // wave 0..11 == m-tile
  int n    = lane & 15;
  int hi   = lane >> 4;

  v8f acc[4] = {};
  for (int ks = 0; ks < kKSpre; ++ks) {
    __syncthreads();
    if (ks < 16) {           // all 32 channels valid: no bound checks
      for (int i = tid; i < 32 * 64; i += 384) {
        int row = i >> 6, col = i & 63;
        float v = x[((size_t)(b * 513 + ks * 32 + row)) * kT + t0 + col];
        xT[col * 40 + row] = (_Float16)v;
      }
    } else {                 // c = 512..543: only c==512 exists
      for (int i = tid; i < 32 * 64; i += 384) {
        int row = i >> 6, col = i & 63;
        float v = (row == 0) ? x[((size_t)(b * 513 + 512)) * kT + t0 + col] : 0.f;
        xT[col * 40 + row] = (_Float16)v;
      }
    }
    __syncthreads();
    const v16h a = *(const v16h*)(fragPre + ((size_t)(w * 17 + ks)) * 512 + lane * 16);
    for (int nt = 0; nt < 4; ++nt) {
      const v8h* p = (const v8h*)(xT + (nt * 16 + n) * 40 + hi * 16);
      v8h lo = p[0], hp = p[1];
      v16h bf = __builtin_shufflevector(lo, hp, 0,1,2,3,4,5,6,7,8,9,10,11,12,13,14,15);
      acc[nt] = WMMA16(a, bf, acc[nt]);
    }
  }

  int len   = xlen[b];
  int cbase = w * 16 + hi * 8;
  for (int nt = 0; nt < 4; ++nt) {
    int t = t0 + nt * 16 + n;
    float mk = (t < len) ? 1.f : 0.f;
    float* hp = h0 + ((size_t)(b * kT + t)) * kH + cbase;
    v8h ph;
    for (int r = 0; r < 8; ++r) {
      float hv = (acc[nt][r] + pre_b[cbase + r]) * mk;
      hp[r] = hv;
      ph[r] = (_Float16)hv;
    }
    *(v8h*)(h16a + ((size_t)b * kTP + t + 2) * kH + cbase) = ph;
  }
  for (int i = tid; i < kTT * kH; i += 384)
    skip[((size_t)(b * kT + t0 + i / kH)) * kH + (i % kH)] = 0.f;

  // zero the 2-row halo pads (front for tile 0, back for last tile)
  if (blockIdx.x == 0) {
    for (int i = tid; i < 2 * kH; i += 384) {
      size_t base = (size_t)b * kTP * kH;
      h16a[base + i] = (_Float16)0.f;
      h16b[base + i] = (_Float16)0.f;
    }
  } else if (blockIdx.x == gridDim.x - 1) {
    for (int i = tid; i < 2 * kH; i += 384) {
      size_t base = ((size_t)b * kTP + kT + 2) * kH;
      h16a[base + i] = (_Float16)0.f;
      h16b[base + i] = (_Float16)0.f;
    }
  }
}

// ---------------------------------------------------------------------------
// One WaveNet layer, fully fused (block = batch x 64-time tile, 12 waves).
// h tile staged by per-wave TDM DMA; layer biases async-copied to LDS.
// Wave w owns m-tiles {w, w+12} so the tanh/sigmoid pairing is lane-local.
// ---------------------------------------------------------------------------
__global__ void __launch_bounds__(384)
wn_layer(const float* __restrict__ hin, float* __restrict__ hout,
         const _Float16* __restrict__ h16in, _Float16* __restrict__ h16out,
         float* __restrict__ skip, const int* __restrict__ xlen,
         const _Float16* __restrict__ fragIn, const _Float16* __restrict__ fragRs,
         const float* __restrict__ in_b, const float* __restrict__ rs_b,
         int layer, int is_last) {
  __shared__ __align__(32) _Float16 sh[68 * kH]; // h tile (+/-2 halo); reused for acts
  __shared__ __align__(16) float sbias[768];     // in_b[layer] (384) | rs_b[layer] (384)
  int b  = blockIdx.y;
  int t0 = blockIdx.x * kTT;
  int tid  = threadIdx.x;
  int lane = tid & 31;
  int w    = tid >> 5;
  int n    = lane & 15;
  int hi   = lane >> 4;

  // TDM: wave w DMAs its contiguous 2176B slice of the 68x192 f16 tile
  {
    const char* gsrc = (const char*)(h16in + ((size_t)b * kTP + t0) * kH);
    unsigned lds0 = (unsigned)(size_t)(void*)sh;
    tdm_copy_1d(lds0 + w * kSliceB, gsrc + w * kSliceB, kSliceU);

    // async-copy this layer's biases (2 x 1536B contiguous) into LDS
    unsigned ldsb = (unsigned)(size_t)(void*)sbias;
    const char* bi = (const char*)(in_b + layer * 384);
    const char* br = (const char*)(rs_b + layer * 384);
    for (int i = tid; i < 96; i += 384) {
      async_b128(ldsb + i * 16,        bi + i * 16);
      async_b128(ldsb + 1536 + i * 16, br + i * 16);
    }
    wait_async0();
    wait_tensor0();
  }
  __syncthreads();

  v8f acc0[4] = {}, acc1[4] = {};
  for (int d = 0; d < 5; ++d) {
    for (int ks = 0; ks < kKS; ++ks) {
      size_t fb = ((size_t)((layer * 5 + d) * 24) + w) * 6 + ks;
      const v16h a0 = *(const v16h*)(fragIn + fb * 512 + lane * 16);
      const v16h a1 = *(const v16h*)(fragIn + (fb + 12 * 6) * 512 + lane * 16);
      for (int nt = 0; nt < 4; ++nt) {
        int lt = nt * 16 + n + d;  // halo-shifted time row
        const v16h bf = *(const v16h*)(sh + lt * kH + ks * 32 + hi * 16);
        acc0[nt] = WMMA16(a0, bf, acc0[nt]);
        acc1[nt] = WMMA16(a1, bf, acc1[nt]);
      }
    }
  }
  __syncthreads();   // all waves done reading h tile

  int len   = xlen[b];
  int cbase = w * 16 + hi * 8;
  // gate: tanh(a)*sigmoid(g) = (e^{2a}-1) / ((e^{2a}+1)*(1+e^{-g}))
  for (int nt = 0; nt < 4; ++nt) {
    int t = t0 + nt * 16 + n;
    float mk = (t < len) ? 1.f : 0.f;
    v8h pk;
    for (int r = 0; r < 8; ++r) {
      float xl = acc0[nt][r] + sbias[cbase + r];
      float xh = acc1[nt][r] + sbias[192 + cbase + r];
      float e2 = __expf(2.f * xl);
      float en = __expf(-xh);
      float a  = mk * (e2 - 1.f) / ((e2 + 1.f) * (1.f + en));
      pk[r] = (_Float16)a;
    }
    *(v8h*)(sh + (nt * 16 + n) * kH + cbase) = pk;   // acts, time-major
  }
  __syncthreads();

  // rs 1x1 conv (last layer: only the low 192 rows)
  for (int nt = 0; nt < 4; ++nt) { acc0[nt] = vzero8(); acc1[nt] = vzero8(); }
  for (int ks = 0; ks < kKS; ++ks) {
    size_t fb = ((size_t)(layer * 24) + w) * 6 + ks;
    const v16h a0 = *(const v16h*)(fragRs + fb * 512 + lane * 16);
    v16h a1 = a0;
    if (!is_last) a1 = *(const v16h*)(fragRs + (fb + 12 * 6) * 512 + lane * 16);
    for (int nt = 0; nt < 4; ++nt) {
      const v16h bf = *(const v16h*)(sh + (nt * 16 + n) * kH + ks * 32 + hi * 16);
      acc0[nt] = WMMA16(a0, bf, acc0[nt]);
      if (!is_last) acc1[nt] = WMMA16(a1, bf, acc1[nt]);
    }
  }

  for (int nt = 0; nt < 4; ++nt) {
    int t = t0 + nt * 16 + n;
    float mk = (t < len) ? 1.f : 0.f;
    size_t base = ((size_t)(b * kT + t)) * kH + cbase;
    if (!is_last) {
      float*       hp = hout + base;
      const float* hq = hin  + base;
      float*       sp = skip + base;
      v8h ph;
      for (int r = 0; r < 8; ++r) {
        float rl = acc0[nt][r] + sbias[384 + cbase + r];
        float hv = (hq[r] + rl) * mk;
        hp[r] = hv;
        ph[r] = (_Float16)hv;
        float rh = acc1[nt][r] + sbias[576 + cbase + r];
        sp[r] += rh;
      }
      *(v8h*)(h16out + ((size_t)b * kTP + t + 2) * kH + cbase) = ph;
    } else {
      float* sp = skip + base;
      for (int r = 0; r < 8; ++r)
        sp[r] += acc0[nt][r] + sbias[384 + cbase + r];
    }
  }
}

// ---------------------------------------------------------------------------
// Post: stats = conv1(skip*mask)*mask ; z = mu + noise*exp(log_s)
// Writes z, mu, log_s (B,192,T each) and mask (B,T).
// ---------------------------------------------------------------------------
__global__ void __launch_bounds__(384)
post_kernel(const float* __restrict__ skip, const float* __restrict__ noise,
            const int* __restrict__ xlen, const _Float16* __restrict__ fragPost,
            const float* __restrict__ post_b, float* __restrict__ out) {
  __shared__ __align__(32) _Float16 sk[kTT * kH];
  int b  = blockIdx.y;
  int t0 = blockIdx.x * kTT;
  int tid  = threadIdx.x;
  int lane = tid & 31;
  int w    = tid >> 5;
  int n    = lane & 15;
  int hi   = lane >> 4;
  int len  = xlen[b];

  for (int i = tid; i < kTT * kH; i += 384) {
    int t = t0 + i / kH;
    float v = (t < len) ? skip[((size_t)(b * kT + t)) * kH + (i % kH)] : 0.f;
    sk[i] = (_Float16)v;
  }
  __syncthreads();

  v8f acc0[4] = {}, acc1[4] = {};
  for (int ks = 0; ks < kKS; ++ks) {
    const v16h a0 = *(const v16h*)(fragPost + ((size_t)(w * 6 + ks)) * 512 + lane * 16);
    const v16h a1 = *(const v16h*)(fragPost + ((size_t)((w + 12) * 6 + ks)) * 512 + lane * 16);
    for (int nt = 0; nt < 4; ++nt) {
      const v16h bf = *(const v16h*)(sk + (nt * 16 + n) * kH + ks * 32 + hi * 16);
      acc0[nt] = WMMA16(a0, bf, acc0[nt]);
      acc1[nt] = WMMA16(a1, bf, acc1[nt]);
    }
  }

  const size_t Z = (size_t)16 * kH * kT;
  int cbase = w * 16 + hi * 8;
  for (int nt = 0; nt < 4; ++nt) {
    int t = t0 + nt * 16 + n;
    float mk = (t < len) ? 1.f : 0.f;
    for (int r = 0; r < 8; ++r) {
      int c = cbase + r;
      float mu = (acc0[nt][r] + post_b[c]) * mk;
      float ls = (acc1[nt][r] + post_b[192 + c]) * mk;
      size_t o = ((size_t)(b * kH + c)) * kT + t;
      out[o]         = mu + noise[o] * __expf(ls);
      out[Z + o]     = mu;
      out[2 * Z + o] = ls;
    }
  }
  if (tid < kTT) {
    int t = t0 + tid;
    out[3 * Z + (size_t)b * kT + t] = (t < len) ? 1.f : 0.f;
  }
}

// ---------------------------------------------------------------------------
extern "C" void kernel_launch(void* const* d_in, const int* in_sizes, int n_in,
                              void* d_out, int out_size, void* d_ws, size_t ws_size,
                              hipStream_t stream) {
  (void)in_sizes; (void)n_in; (void)out_size; (void)ws_size;
  const float* x      = (const float*)d_in[0];
  const int*   xlen   = (const int*)d_in[1];
  const float* noise  = (const float*)d_in[2];
  const float* pre_w  = (const float*)d_in[3];
  const float* pre_b  = (const float*)d_in[4];
  const float* in_w   = (const float*)d_in[5];
  const float* in_b   = (const float*)d_in[6];
  const float* rs_w   = (const float*)d_in[7];
  const float* rs_b   = (const float*)d_in[8];
  const float* post_w = (const float*)d_in[9];
  const float* post_b = (const float*)d_in[10];
  float* out = (float*)d_out;

  char* ws = (char*)d_ws;
  size_t off = 0;
  auto carve = [&](size_t bytes) {
    void* p = ws + off;
    off = (off + bytes + 255) & ~(size_t)255;
    return p;
  };
  const size_t hBytes   = (size_t)16 * kT * kH * sizeof(float);      // 25.2 MB
  const size_t h16Bytes = (size_t)16 * kTP * kH * sizeof(_Float16);  // 12.6 MB
  float*    hA       = (float*)carve(hBytes);
  float*    hB       = (float*)carve(hBytes);
  float*    skipBuf  = (float*)carve(hBytes);
  _Float16* h16A     = (_Float16*)carve(h16Bytes);
  _Float16* h16B     = (_Float16*)carve(h16Bytes);
  _Float16* fragIn   = (_Float16*)carve((size_t)16 * 5 * 24 * 6 * 512 * 2);
  _Float16* fragRs   = (_Float16*)carve((size_t)16 * 24 * 6 * 512 * 2);
  _Float16* fragPost = (_Float16*)carve((size_t)24 * 6 * 512 * 2);
  _Float16* fragPre  = (_Float16*)carve((size_t)12 * 17 * 512 * 2);

  prep_frags<<<16 * 5 * 24 * 6, 512, 0, stream>>>(in_w,   fragIn,   0);
  prep_frags<<<16 * 24 * 6,     512, 0, stream>>>(rs_w,   fragRs,   1);
  prep_frags<<<24 * 6,          512, 0, stream>>>(post_w, fragPost, 2);
  prep_frags<<<12 * 17,         512, 0, stream>>>(pre_w,  fragPre,  3);

  dim3 grid(kT / kTT, 16);   // (32 time tiles, B)
  pre_conv<<<grid, 384, 0, stream>>>(x, xlen, fragPre, pre_b, hA, skipBuf, h16A, h16B);

  for (int l = 0; l < 16; ++l) {
    float*    hi_  = (l & 1) ? hB : hA;
    float*    ho_  = (l & 1) ? hA : hB;
    _Float16* h16i = (l & 1) ? h16B : h16A;
    _Float16* h16o = (l & 1) ? h16A : h16B;
    wn_layer<<<grid, 384, 0, stream>>>(hi_, ho_, h16i, h16o, skipBuf, xlen,
                                       fragIn, fragRs, in_b, rs_b, l, (l == 15) ? 1 : 0);
  }

  post_kernel<<<grid, 384, 0, stream>>>(skipBuf, noise, xlen, fragPost, post_b, out);
}
